// DGMGMinimal_86311662780917
// MI455X (gfx1250) — matrix operand order repaired and checked by reading, not compiled, not run on hardware
//
#include <hip/hip_runtime.h>
#include <math.h>

#define Hdim 512
#define NMAX 128
#define LSEQ 1024

typedef __attribute__((ext_vector_type(16))) __bf16 v16bf;
typedef __attribute__((ext_vector_type(8)))  __bf16 v8bf;
typedef __attribute__((ext_vector_type(8)))  float  v8f;

// ---------------- workspace layout (bytes, all offsets 32B-aligned) ----------------
static constexpr size_t SZ_GRUW  = (size_t)3 * Hdim * Hdim * 2;            // 1.5 MB
static constexpr size_t OFF_WIHN = 0;
static constexpr size_t OFF_WHHN = OFF_WIHN + SZ_GRUW;
static constexpr size_t OFF_WIHG = OFF_WHHN + SZ_GRUW;
static constexpr size_t OFF_WHHG = OFF_WIHG + SZ_GRUW;
static constexpr size_t OFF_WE1  = OFF_WHHG + SZ_GRUW;                     // 512x1024 bf16
static constexpr size_t OFF_WQ   = OFF_WE1 + (size_t)Hdim * 2 * Hdim * 2;  // 512x1024 bf16
static constexpr size_t OFF_WK   = OFF_WQ  + (size_t)Hdim * 2 * Hdim * 2;  // 512x512  bf16
static constexpr size_t OFF_NEF  = OFF_WK  + (size_t)Hdim * Hdim * 2;      // ne  f32
static constexpr size_t OFF_NEBF = OFF_NEF + (size_t)NMAX * Hdim * 4;      // ne  bf16
static constexpr size_t OFF_KC   = OFF_NEBF+ (size_t)NMAX * Hdim * 2;      // K-cache f32
static constexpr size_t OFF_H0   = OFF_KC  + (size_t)NMAX * Hdim * 4;      // h0  f32
static constexpr size_t OFF_H0BF = OFF_H0  + (size_t)Hdim * 4;             // h0  bf16
static constexpr size_t OFF_GI0  = OFF_H0BF+ (size_t)Hdim * 2;             // wih_g@h0+bih_g f32
static constexpr size_t OFF_K0   = OFF_GI0 + (size_t)3 * Hdim * 4;         // w_k@h0+b_k f32

// ---------------- small helpers ----------------
__device__ __forceinline__ float waveReduce(float v) {
#pragma unroll
  for (int o = 16; o > 0; o >>= 1) v += __shfl_xor(v, o, 32);
  return v;
}

__device__ __forceinline__ float gru_out(int j, const float* gi, const float* gh, float hprev) {
  float r = 1.0f / (1.0f + __expf(-(gi[j]        + gh[j])));
  float z = 1.0f / (1.0f + __expf(-(gi[Hdim + j] + gh[Hdim + j])));
  float n = tanhf(gi[2 * Hdim + j] + r * gh[2 * Hdim + j]);
  return (1.0f - z) * n + z * hprev;
}

// WMMA mat-vec: out[0..M) = W(MxK,bf16,row-major) @ xbf(K) + bias.
//
// Activation chunk rides in row 0 of the A fragment. With the contiguous k order used here,
// row M=0 is sourced from lane 0 (K 0-15) and lane 16 (K 16-31).
// Branch-free trick: ALL lanes load the same pattern (lanes >=16 shifted by 16
// elements). Rows M=1..15 of A get harmless copies of x — they only feed D
// rows that are never read (we read only D[0][n] = acc[0], lanes 0-15), and
// WMMA raises no exceptions. This keeps the inner loop at 2 ds_load_b128 +
// 2 global_load_b128 + 1 v_wmma, with no exec-mask manipulation.
//
// B tile: lane n / n+16 hold the two 16-element k-halves of weight row rb+n:
// coalesced 32B/lane global_load_b128 pairs, each weight byte used once.
__device__ __forceinline__ void matvec_bf16(const __bf16* __restrict__ W,
                                            const __bf16* xbf, int M, int K,
                                            const float* __restrict__ bias,
                                            float* out) {
  const int tid  = threadIdx.x;
  const int lane = tid & 31;
  const int wave = tid >> 5;
  const int nt   = M >> 4;                   // 16-row tiles
  const int n    = lane & 15;
  const int hi   = lane >> 4;                // 0 or 1 (K-half)
  const __bf16* xlane = xbf + (hi << 4);     // +16 elements for high-half lanes

  for (int t = wave; t < nt; t += 16) {
    const int rb = t << 4;
    const __bf16* wrow = W + (size_t)(rb + n) * K + (hi << 4);
    v8f acc = {};
#pragma unroll 4
    for (int k0 = 0; k0 < K; k0 += 32) {
      v8bf alo = *reinterpret_cast<const v8bf*>(xlane + k0);        // K 0-7 (of chunk half)
      v8bf ahi = *reinterpret_cast<const v8bf*>(xlane + k0 + 8);    // K 8-15 (of chunk half)
      v16bf a;
#pragma unroll
      for (int j = 0; j < 8; ++j) { a[j] = alo[j]; a[8 + j] = ahi[j]; }
      v16bf b = *reinterpret_cast<const v16bf*>(wrow + k0);
      acc = __builtin_amdgcn_wmma_f32_16x16x32_bf16(false, a, false, b,
                                                    (short)0, acc, false, false);
    }
    if (lane < 16) out[rb + lane] = acc[0] + bias[rb + lane];
  }
}

// ---------------- setup kernels ----------------
__global__ void convert_bf16(const float* __restrict__ src, void* dstv, int nelem) {
  __bf16* dst = (__bf16*)dstv;
  int i = blockIdx.x * blockDim.x + threadIdx.x;
  if (i < nelem) dst[i] = (__bf16)src[i];
}

__global__ void zero_state(float* neF, void* neBFv, float* Kc, int nelem) {
  __bf16* neBF = (__bf16*)neBFv;
  int i = blockIdx.x * blockDim.x + threadIdx.x;
  if (i < nelem) { neF[i] = 0.0f; neBF[i] = (__bf16)0.0f; Kc[i] = 0.0f; }
}

// h0 = w_nip@node_init + b_nip ; gi0 = wih_g@h0 + bih_g ; k0 = w_k@h0 + b_k (all fp32)
__global__ void __launch_bounds__(Hdim) precompute(
    const float* __restrict__ node_init, const float* __restrict__ w_nip,
    const float* __restrict__ b_nip, const float* __restrict__ wih_g,
    const float* __restrict__ bih_g, const float* __restrict__ w_k,
    const float* __restrict__ b_k, float* h0, void* h0bfv, float* gi0, float* k0) {
  __shared__ float s_h0[Hdim];
  __bf16* h0bf = (__bf16*)h0bfv;
  const int tid = threadIdx.x;
  float acc = b_nip[tid];
  for (int j = 0; j < Hdim; ++j) acc += w_nip[tid * Hdim + j] * node_init[j];
  s_h0[tid] = acc; h0[tid] = acc; h0bf[tid] = (__bf16)acc;
  __syncthreads();
  for (int m = tid; m < 3 * Hdim; m += Hdim) {
    float a = bih_g[m];
    for (int j = 0; j < Hdim; ++j) a += wih_g[(size_t)m * Hdim + j] * s_h0[j];
    gi0[m] = a;
  }
  float a = b_k[tid];
  for (int j = 0; j < Hdim; ++j) a += w_k[tid * Hdim + j] * s_h0[j];
  k0[tid] = a;
}

// ---------------- persistent sequential scan (1 workgroup, 16 waves) ----------------
__global__ void __launch_bounds__(Hdim) dgmg_scan(
    const int* __restrict__ actions, const int* __restrict__ argseq,
    const float* __restrict__ graph0, const float* __restrict__ b_e1,
    const float* __restrict__ w_e2, const float* __restrict__ b_e2,
    const float* __restrict__ b_q, const float* __restrict__ b_k,
    const float* __restrict__ bih_n, const float* __restrict__ bhh_n,
    const float* __restrict__ bih_g, const float* __restrict__ bhh_g,
    float* __restrict__ out, char* __restrict__ ws) {
  const __bf16* WIHN = (const __bf16*)(ws + OFF_WIHN);
  const __bf16* WHHN = (const __bf16*)(ws + OFF_WHHN);
  const __bf16* WIHG = (const __bf16*)(ws + OFF_WIHG);
  const __bf16* WHHG = (const __bf16*)(ws + OFF_WHHG);
  const __bf16* WE1  = (const __bf16*)(ws + OFF_WE1);
  const __bf16* WQ   = (const __bf16*)(ws + OFF_WQ);
  const __bf16* WK   = (const __bf16*)(ws + OFF_WK);
  float*  neF  = (float*)(ws + OFF_NEF);
  __bf16* neBF = (__bf16*)(ws + OFF_NEBF);
  float*  Kc   = (float*)(ws + OFF_KC);
  const float*  h0   = (const float*)(ws + OFF_H0);
  const __bf16* h0bf = (const __bf16*)(ws + OFF_H0BF);
  const float*  gi0  = (const float*)(ws + OFF_GI0);
  const float*  k0v  = (const float*)(ws + OFF_K0);

  __shared__ float  s_g[Hdim], s_curv[Hdim], s_gi[3 * Hdim], s_gh[3 * Hdim], s_vec[Hdim];
  __shared__ __bf16 s_gbf[Hdim], s_curvbf[Hdim], s_x[2 * Hdim];
  __shared__ float  s_logits[NMAX];
  __shared__ float  s_red[2];
  __shared__ float  s_nll;
  __shared__ int    s_cur;

  const int tid  = threadIdx.x;
  const int lane = tid & 31;
  const int wave = tid >> 5;

  { float v = graph0[tid]; s_g[tid] = v; s_gbf[tid] = (__bf16)v; }
  s_curv[tid] = 0.0f; s_curvbf[tid] = (__bf16)0.0f;
  if (tid == 0) { s_nll = 0.0f; s_cur = -1; }
  __syncthreads();

#pragma unroll 1
  for (int t = 0; t < LSEQ; ++t) {
    const int a   = actions[t];
    const int arg = argseq[t];

    if (a == 0) {                                   // ---- ADD_NODE ----
      const int cur2 = s_cur + 1;
      {
        float v = h0[tid]; __bf16 vb = h0bf[tid];
        neF[(size_t)cur2 * Hdim + tid] = v;  neBF[(size_t)cur2 * Hdim + tid] = vb;
        s_curv[tid] = v;  s_curvbf[tid] = vb;
        Kc[(size_t)cur2 * Hdim + tid] = k0v[tid];
        s_gi[tid] = gi0[tid]; s_gi[Hdim + tid] = gi0[Hdim + tid];
        s_gi[2 * Hdim + tid] = gi0[2 * Hdim + tid];
      }
      __threadfence();
      __syncthreads();
      if (tid == 0) s_cur = cur2;
      matvec_bf16(WHHG, s_gbf, 3 * Hdim, Hdim, bhh_g, s_gh);  // gh = whh_g@g + bhh_g
      __syncthreads();
      { float hn = gru_out(tid, s_gi, s_gh, s_g[tid]);
        s_g[tid] = hn; s_gbf[tid] = (__bf16)hn; }
      __syncthreads();

    } else if (a == 1 || a == 3) {                  // ---- ADD_EDGE / STOP_EDGE ----
      s_x[tid] = s_gbf[tid]; s_x[Hdim + tid] = s_curvbf[tid];
      __syncthreads();
      matvec_bf16(WE1, s_x, Hdim, 2 * Hdim, b_e1, s_vec);
      __syncthreads();
      s_vec[tid] = fmaxf(s_vec[tid], 0.0f);         // relu
      __syncthreads();
      if (wave < 2) {                               // 2-row head w_e2 (2x512)
        float p = 0.0f;
        for (int j = lane; j < Hdim; j += 32) p += w_e2[wave * Hdim + j] * s_vec[j];
        p = waveReduce(p);
        if (lane == 0) s_red[wave] = p + b_e2[wave];
      }
      __syncthreads();
      if (tid == 0) {
        float l0 = s_red[0], l1 = s_red[1];
        float m = fmaxf(l0, l1);
        float lse = m + __logf(__expf(l0 - m) + __expf(l1 - m));
        s_nll += lse - ((a == 1) ? l0 : l1);
      }
      __syncthreads();

    } else if (a == 2) {                            // ---- CHOOSE_DEST ----
      const int cur = s_cur;
      s_x[tid] = s_gbf[tid]; s_x[Hdim + tid] = s_curvbf[tid];
      __syncthreads();
      matvec_bf16(WQ, s_x, Hdim, 2 * Hdim, b_q, s_vec);       // q
      __syncthreads();
      for (int i = wave; i < cur; i += 16) {        // logits from incremental K-cache
        float p = 0.0f;
        for (int j = lane; j < Hdim; j += 32) p += Kc[(size_t)i * Hdim + j] * s_vec[j];
        p = waveReduce(p);
        if (lane == 0) s_logits[i] = p;
      }
      __syncthreads();
      if (tid == 0) {
        float m = -3.0e38f;
        for (int i = 0; i < cur; ++i) m = fmaxf(m, s_logits[i]);
        float ssum = 0.0f;
        for (int i = 0; i < cur; ++i) ssum += __expf(s_logits[i] - m);
        s_nll += (m + __logf(ssum)) - s_logits[arg];
      }
      s_x[tid] = neBF[(size_t)arg * Hdim + tid];    // chosen node (bf16)
      __syncthreads();
      matvec_bf16(WIHN, s_x,      3 * Hdim, Hdim, bih_n, s_gi);
      matvec_bf16(WHHN, s_curvbf, 3 * Hdim, Hdim, bhh_n, s_gh);
      __syncthreads();
      { float hn = gru_out(tid, s_gi, s_gh, s_curv[tid]);     // new current-node vec
        __bf16 hb = (__bf16)hn;
        s_curv[tid] = hn; s_curvbf[tid] = hb;
        neF[(size_t)cur * Hdim + tid] = hn; neBF[(size_t)cur * Hdim + tid] = hb; }
      __syncthreads();
      matvec_bf16(WK,   s_curvbf, Hdim,     Hdim, b_k,   Kc + (size_t)cur * Hdim);
      matvec_bf16(WIHG, s_curvbf, 3 * Hdim, Hdim, bih_g, s_gi);
      matvec_bf16(WHHG, s_gbf,    3 * Hdim, Hdim, bhh_g, s_gh);
      __threadfence();
      __syncthreads();
      { float hn = gru_out(tid, s_gi, s_gh, s_g[tid]);        // graph GRU
        s_g[tid] = hn; s_gbf[tid] = (__bf16)hn; }
      __syncthreads();
    }
    // a<0 or a>=4: noop
  }
  if (tid == 0) out[0] = s_nll;
}

// ---------------- launch ----------------
extern "C" void kernel_launch(void* const* d_in, const int* in_sizes, int n_in,
                              void* d_out, int out_size, void* d_ws, size_t ws_size,
                              hipStream_t stream) {
  (void)in_sizes; (void)n_in; (void)out_size; (void)ws_size;
  const int*   actions   = (const int*)d_in[0];
  const int*   argseq    = (const int*)d_in[1];
  const float* node_init = (const float*)d_in[2];
  const float* w_nip     = (const float*)d_in[3];
  const float* b_nip     = (const float*)d_in[4];
  const float* w_e1      = (const float*)d_in[5];
  const float* b_e1      = (const float*)d_in[6];
  const float* w_e2      = (const float*)d_in[7];
  const float* b_e2      = (const float*)d_in[8];
  const float* w_q       = (const float*)d_in[9];
  const float* b_q       = (const float*)d_in[10];
  const float* w_k       = (const float*)d_in[11];
  const float* b_k       = (const float*)d_in[12];
  const float* wih_n     = (const float*)d_in[13];
  const float* whh_n     = (const float*)d_in[14];
  const float* bih_n     = (const float*)d_in[15];
  const float* bhh_n     = (const float*)d_in[16];
  const float* wih_g     = (const float*)d_in[17];
  const float* whh_g     = (const float*)d_in[18];
  const float* bih_g     = (const float*)d_in[19];
  const float* bhh_g     = (const float*)d_in[20];
  const float* graph0    = (const float*)d_in[21];
  char* ws = (char*)d_ws;

  auto cvt = [&](const float* src, size_t off, int nelem) {
    convert_bf16<<<(nelem + 255) / 256, 256, 0, stream>>>(src, (void*)(ws + off), nelem);
  };
  cvt(wih_n, OFF_WIHN, 3 * Hdim * Hdim);
  cvt(whh_n, OFF_WHHN, 3 * Hdim * Hdim);
  cvt(wih_g, OFF_WIHG, 3 * Hdim * Hdim);
  cvt(whh_g, OFF_WHHG, 3 * Hdim * Hdim);
  cvt(w_e1,  OFF_WE1,  Hdim * 2 * Hdim);
  cvt(w_q,   OFF_WQ,   Hdim * 2 * Hdim);
  cvt(w_k,   OFF_WK,   Hdim * Hdim);

  const int nstate = NMAX * Hdim;
  zero_state<<<(nstate + 255) / 256, 256, 0, stream>>>(
      (float*)(ws + OFF_NEF), (void*)(ws + OFF_NEBF), (float*)(ws + OFF_KC), nstate);

  precompute<<<1, Hdim, 0, stream>>>(node_init, w_nip, b_nip, wih_g, bih_g, w_k, b_k,
                                     (float*)(ws + OFF_H0), (void*)(ws + OFF_H0BF),
                                     (float*)(ws + OFF_GI0), (float*)(ws + OFF_K0));

  dgmg_scan<<<1, Hdim, 0, stream>>>(actions, argseq, graph0, b_e1, w_e2, b_e2, b_q, b_k,
                                    bih_n, bhh_n, bih_g, bhh_g, (float*)d_out, ws);
}
